// Mamba2LiteMixer_30923764531649
// MI455X (gfx1250) — matrix operand order repaired and checked by reading, not compiled, not run on hardware
//
#include <hip/hip_runtime.h>
#include <hip/hip_bf16.h>

typedef __attribute__((ext_vector_type(16))) __bf16 v16bf;
typedef __attribute__((ext_vector_type(8)))  __bf16 v8bf;
typedef __attribute__((ext_vector_type(8)))  float  v8f;
typedef __attribute__((ext_vector_type(4)))  unsigned v4u;
typedef __attribute__((ext_vector_type(8)))  int      v8i;
typedef __attribute__((ext_vector_type(4)))  int      v4i;

#define D_MODEL 1024
#define D_INNER 2048
#define BATCH   4
#define SEQ     4096
#define MTOT    (BATCH * SEQ)   // 16384
#define NC      8               // scan chunks per sequence
#define CL      (SEQ / NC)      // 512

// ---------------- TDM: 2D tile load Global -> LDS --------------------------
// Loads a tile of tile_y rows x tile_x bf16 elements, row stride = stride_elems,
// into LDS at lds_off, padding +16B after every 64B (row stride 80B in LDS,
// matching the 40-half padded layout the WMMA fragment reads expect).
// Descriptor packing per CDNA5 ISA section 8.3/8.4 (D# group0/group1).
__device__ __forceinline__ void tdm_load_tile_bf16(
    unsigned lds_off, const __bf16* gptr, unsigned tile_x, unsigned tile_y,
    unsigned long long stride_elems, unsigned tensor_d0, unsigned tensor_d1)
{
    unsigned long long ga = (unsigned long long)(const void*)gptr;
    v4u g0;
    g0.x = 1u;                                          // count=1, is_restore=0
    g0.y = lds_off;                                     // lds_addr [63:32]
    g0.z = (unsigned)(ga & 0xffffffffu);                // global_addr lo
    g0.w = (unsigned)((ga >> 32) & 0x1ffffffu)          // global_addr hi (57-bit)
         | (2u << 30);                                  // type=2 ("image")
    v8i g1;
    // data_size=1 (2 bytes), pad_enable=1, pad_interval=3 (16 DW = 64B),
    // pad_amount=3 (4 DW = 16B); workgroup_mask=0
    g1[0] = (int)((1u << 16) | (1u << 20) | (3u << 22) | (3u << 25));
    g1[1] = (int)((tensor_d0 & 0xffffu) << 16);         // tensor_dim0[15:0] @63:48
    g1[2] = (int)((tensor_d0 >> 16) | ((tensor_d1 & 0xffffu) << 16));
    g1[3] = (int)((tensor_d1 >> 16) | (tile_x << 16));  // tile_dim0 @127:112
    g1[4] = (int)(tile_y & 0xffffu);                    // tile_dim1, tile_dim2=0
    g1[5] = (int)(unsigned)(stride_elems & 0xffffffffull);   // dim0_stride lo32
    g1[6] = (int)(unsigned)((stride_elems >> 32) & 0xffffull); // dim0_stride hi16
    g1[7] = 0;                                          // tensor_dim1_stride=0 (2D)
    v4i gz = {0, 0, 0, 0};
#if __clang_major__ >= 23
    v8i gz8 = {0, 0, 0, 0, 0, 0, 0, 0};
    __builtin_amdgcn_tensor_load_to_lds(g0, g1, gz, gz, gz8, 0);
#else
    __builtin_amdgcn_tensor_load_to_lds(g0, g1, gz, gz, 0);
#endif
}

// ---------------- f32 -> bf16 conversion ----------------
__global__ void cvt_bf16_kernel(const float* __restrict__ s, __bf16* __restrict__ d, int n) {
    int i = blockIdx.x * blockDim.x + threadIdx.x;
    if (i < n) d[i] = (__bf16)s[i];
}

// ---------------- WMMA GEMM: out[m,n] = sum_k A[m,k] * W[n,k] (+epilogue) ----
// A: bf16 [M,K] row-major, W: bf16 [N,K] row-major. M%128==0, N%128==0, K%32==0.
// Block tile 128x128, BK=32, 256 threads = 8 waves (4 along M x 2 along N),
// wave tile 32x64 = 2x4 WMMA 16x16 tiles, f32 accumulation.
// Tile staging: TDM tensor_load_to_lds, double-buffered, issued by wave 0.
template <int EPI>
__global__ __launch_bounds__(256)
void gemm_wmma(const __bf16* __restrict__ A, const __bf16* __restrict__ W,
               int K, int N,
               const float* __restrict__ bias,
               const float* __restrict__ dt_b, const float* __restrict__ A_log,
               __bf16* __restrict__ outH, float* __restrict__ outF,
               float* __restrict__ decayOut,
               __bf16* __restrict__ BhOut, __bf16* __restrict__ ChOut)
{
    constexpr int LDSS = 40;  // bf16 elems per LDS row: 32 data + 8 pad (TDM pads 16B/64B)
    __shared__ __align__(16) __bf16 As[2][128 * LDSS];
    __shared__ __align__(16) __bf16 Bs[2][128 * LDSS];

    const int tid   = threadIdx.x;
    const int lane  = tid & 31;
    const int wave  = tid >> 5;
    const int waveM = wave & 3;   // 0..3
    const int waveN = wave >> 2;  // 0..1
    const int half  = lane >> 4;  // 0|1
    const int l16   = lane & 15;
    const int blockN0 = blockIdx.x * 128;
    const int blockM0 = blockIdx.y * 128;

    v8f acc[2][4];
#pragma unroll
    for (int i = 0; i < 2; ++i)
#pragma unroll
        for (int j = 0; j < 4; ++j) acc[i][j] = (v8f)(0.0f);

    const int kTiles = K >> 5;   // K/32

    if (wave == 0) {  // prologue: stage k-tile 0 into buffer 0
        tdm_load_tile_bf16((unsigned)(uintptr_t)&As[0][0],
                           A + (size_t)blockM0 * K, 32, 128,
                           (unsigned long long)K, (unsigned)K, 0x7fffffffu);
        tdm_load_tile_bf16((unsigned)(uintptr_t)&Bs[0][0],
                           W + (size_t)blockN0 * K, 32, 128,
                           (unsigned long long)K, (unsigned)K, 0x7fffffffu);
    }

    for (int kt = 0; kt < kTiles; ++kt) {
        const int cur = kt & 1;
        if (wave == 0) {
            if (kt + 1 < kTiles) {
                const int k0n = (kt + 1) << 5;
                tdm_load_tile_bf16((unsigned)(uintptr_t)&As[cur ^ 1][0],
                                   A + (size_t)blockM0 * K + k0n, 32, 128,
                                   (unsigned long long)K, (unsigned)K, 0x7fffffffu);
                tdm_load_tile_bf16((unsigned)(uintptr_t)&Bs[cur ^ 1][0],
                                   W + (size_t)blockN0 * K + k0n, 32, 128,
                                   (unsigned long long)K, (unsigned)K, 0x7fffffffu);
                // same-wave tensor ops complete in order: oldest 2 (= current
                // buffer's A+B) are done once <=2 remain outstanding
                __builtin_amdgcn_s_wait_tensorcnt(2);
            } else {
                __builtin_amdgcn_s_wait_tensorcnt(0);
            }
        }
        __syncthreads();   // current buffer visible to all waves

        // A fragment (16x32, MxK): lanes 0-15 hold K 0-7 & 16-23 of row M=lane,
        // lanes 16-31 hold K 8-15 & 24-31 of row M=lane-16.
        v16bf afrag[2];
#pragma unroll
        for (int mt = 0; mt < 2; ++mt) {
            const __bf16* r = &As[cur][(waveM * 32 + mt * 16 + l16) * LDSS];
            v8bf c1 = *(const v8bf*)(r + half * 8);
            v8bf c2 = *(const v8bf*)(r + half * 8 + 16);
            afrag[mt] = __builtin_shufflevector(c1, c2, 0,1,2,3,4,5,6,7,8,9,10,11,12,13,14,15);
        }
        // B fragment (32x16, KxN): lanes 0-15 hold K 0-7 & 16-23 of col N=lane,
        // lanes 16-31 hold K 8-15 & 24-31 of col N=lane-16. W row n supplies col n.
        v16bf bfrag[4];
#pragma unroll
        for (int nt = 0; nt < 4; ++nt) {
            const __bf16* r = &Bs[cur][(waveN * 64 + nt * 16 + l16) * LDSS];
            v8bf c1 = *(const v8bf*)(r + half * 8);
            v8bf c2 = *(const v8bf*)(r + half * 8 + 16);
            bfrag[nt] = __builtin_shufflevector(c1, c2, 0,1,2,3,4,5,6,7,8,9,10,11,12,13,14,15);
        }
#pragma unroll
        for (int mt = 0; mt < 2; ++mt)
#pragma unroll
            for (int nt = 0; nt < 4; ++nt)
                acc[mt][nt] = __builtin_amdgcn_wmma_f32_16x16x32_bf16(
                    false, afrag[mt], false, bfrag[nt],
                    (short)0, acc[mt][nt], false, false);

        __syncthreads();   // all waves done reading before wave0 overwrites
    }

    // epilogue: C/D layout: VGPR j, lanes0-15 -> (M=j, N=lane); lanes16-31 -> (M=j+8, N=lane-16)
#pragma unroll
    for (int mt = 0; mt < 2; ++mt) {
#pragma unroll
        for (int nt = 0; nt < 4; ++nt) {
            const int n     = blockN0 + waveN * 64 + nt * 16 + l16;
            const int mBase = blockM0 + waveM * 32 + mt * 16 + half * 8;
#pragma unroll
            for (int j = 0; j < 8; ++j) {
                const int m = mBase + j;
                float v = acc[mt][nt][j];
                if (EPI == 0) {                       // in_proj: bias + bf16 store
                    outH[(size_t)m * N + n] = (__bf16)(v + bias[n]);
                } else if (EPI == 2) {                // out_proj: bias + f32 store
                    outF[(size_t)m * N + n] = v + bias[n];
                } else {                              // fused dt/B/C epilogue
                    const int seg = n >> 11;          // uniform per block (128 | 2048)
                    const int nn  = n & (D_INNER - 1);
                    if (seg == 0) {
                        float x  = v + dt_b[nn];
                        float sp = (x > 20.0f) ? x : log1pf(expf(x));
                        float dt = sp + 1e-4f;
                        decayOut[(size_t)m * D_INNER + nn] = expf(-dt * expf(A_log[nn]));
                    } else if (seg == 1) {
                        BhOut[(size_t)m * D_INNER + nn] = (__bf16)tanhf(v);
                    } else {
                        ChOut[(size_t)m * D_INNER + nn] = (__bf16)tanhf(v);
                    }
                }
            }
        }
    }
}

// ---------------- depthwise causal conv (k=4) + SiLU ----------------
__global__ __launch_bounds__(256)
void conv_silu_kernel(const __bf16* __restrict__ xzh, const float* __restrict__ cw,
                      const float* __restrict__ cb, __bf16* __restrict__ uh)
{
    const int d = blockIdx.x * blockDim.x + threadIdx.x;  // 0..2047
    const int m = blockIdx.y;                              // 0..MTOT-1
    const int t = m & (SEQ - 1);
    float a = cb[d];
#pragma unroll
    for (int j = 0; j < 4; ++j) {
        int tt = t - 3 + j;
        if (tt >= 0)
            a += cw[d * 4 + j] * (float)xzh[(size_t)(m - 3 + j) * (2 * D_INNER) + d];
    }
    float u = a / (1.0f + expf(-a));   // SiLU
    uh[(size_t)m * D_INNER + d] = (__bf16)u;
}

// ---------------- chunked linear-recurrence scan ----------------
// pass1: per (b, chunk, d) compute P = prod(decay), Q = local scan tail
__global__ __launch_bounds__(256)
void scan_pass1(const float* __restrict__ decay, const __bf16* __restrict__ Bh,
                const __bf16* __restrict__ uh,
                float* __restrict__ P, float* __restrict__ Q)
{
    const int idx = blockIdx.x * blockDim.x + threadIdx.x;   // BATCH*NC*D_INNER
    const int d = idx & (D_INNER - 1);
    const int c = (idx >> 11) & (NC - 1);
    const int b = idx >> 14;
    size_t base = ((size_t)b * SEQ + (size_t)c * CL) * D_INNER + d;
    float p = 1.0f, q = 0.0f;
    for (int t = 0; t < CL; ++t) {
        size_t o = base + (size_t)t * D_INNER;
        __builtin_prefetch(&decay[o + 16 * D_INNER], 0, 0);
        float dec = decay[o];
        float inc = (1.0f - dec) * (float)Bh[o] * (float)uh[o];
        q = dec * q + inc;
        p *= dec;
    }
    P[idx] = p;
    Q[idx] = q;
}

// pass2: serial combine across NC chunks -> chunk-entry states S0
__global__ __launch_bounds__(256)
void scan_pass2(const float* __restrict__ P, const float* __restrict__ Q,
                float* __restrict__ S0)
{
    const int idx = blockIdx.x * blockDim.x + threadIdx.x;   // BATCH*D_INNER
    const int d = idx & (D_INNER - 1);
    const int b = idx >> 11;
    float s = 0.0f;
#pragma unroll
    for (int c = 0; c < NC; ++c) {
        int o = (((b << 3) + c) << 11) + d;
        S0[o] = s;
        s = P[o] * s + Q[o];
    }
}

// pass3: replay chunk with known entry state, fuse y = C*s + D*u, sigmoid(z) gate
__global__ __launch_bounds__(256)
void scan_pass3(const float* __restrict__ decay, const __bf16* __restrict__ Bh,
                const __bf16* __restrict__ uh, const __bf16* __restrict__ Ch,
                const __bf16* __restrict__ xzh, const float* __restrict__ Dp,
                const float* __restrict__ S0, __bf16* __restrict__ yh)
{
    const int idx = blockIdx.x * blockDim.x + threadIdx.x;
    const int d = idx & (D_INNER - 1);
    const int c = (idx >> 11) & (NC - 1);
    const int b = idx >> 14;
    float s = S0[idx];
    const float Dpd = Dp[d];
    size_t base  = ((size_t)b * SEQ + (size_t)c * CL) * D_INNER + d;
    size_t zbase = ((size_t)b * SEQ + (size_t)c * CL) * (2 * D_INNER) + D_INNER + d;
    for (int t = 0; t < CL; ++t) {
        size_t o = base + (size_t)t * D_INNER;
        __builtin_prefetch(&decay[o + 16 * D_INNER], 0, 0);
        float dec = decay[o];
        float u   = (float)uh[o];
        float inc = (1.0f - dec) * (float)Bh[o] * u;
        s = dec * s + inc;
        float y = (float)Ch[o] * s + Dpd * u;
        float z = (float)xzh[zbase + (size_t)t * (2 * D_INNER)];
        y *= 1.0f / (1.0f + expf(-z));
        yh[o] = (__bf16)y;
    }
}

// ---------------- launcher ----------------
extern "C" void kernel_launch(void* const* d_in, const int* in_sizes, int n_in,
                              void* d_out, int out_size, void* d_ws, size_t ws_size,
                              hipStream_t stream)
{
    const float* x      = (const float*)d_in[0];
    const float* w_in   = (const float*)d_in[1];
    const float* b_in   = (const float*)d_in[2];
    const float* conv_w = (const float*)d_in[3];
    const float* conv_b = (const float*)d_in[4];
    const float* dt_w   = (const float*)d_in[5];
    const float* dt_b   = (const float*)d_in[6];
    const float* b_w    = (const float*)d_in[7];
    const float* c_w    = (const float*)d_in[8];
    const float* A_log  = (const float*)d_in[9];
    const float* D_par  = (const float*)d_in[10];
    const float* out_w  = (const float*)d_in[11];
    const float* out_b  = (const float*)d_in[12];
    float* out = (float*)d_out;

    char* wsp = (char*)d_ws;
    auto carve = [&](size_t bytes) {
        char* p = wsp;
        wsp += (bytes + 255) & ~(size_t)255;
        return p;
    };
    __bf16* xh    = (__bf16*)carve((size_t)MTOT * D_MODEL * 2);        // x bf16
    __bf16* w1h   = (__bf16*)carve((size_t)2 * D_INNER * D_MODEL * 2); // in_proj_w bf16
    __bf16* wdh   = (__bf16*)carve((size_t)3 * D_INNER * D_INNER * 2); // [dt;b;c] bf16
    __bf16* woh   = (__bf16*)carve((size_t)D_MODEL * D_INNER * 2);     // out_w bf16
    __bf16* xzh   = (__bf16*)carve((size_t)MTOT * 2 * D_INNER * 2);    // xz bf16
    __bf16* uh    = (__bf16*)carve((size_t)MTOT * D_INNER * 2);        // conv+silu bf16
    float*  decayB= (float*)carve((size_t)MTOT * D_INNER * 4);         // exp(-dt*A)
    __bf16* Bh    = (__bf16*)carve((size_t)MTOT * D_INNER * 2);        // tanh(B)
    __bf16* Ch    = (__bf16*)carve((size_t)MTOT * D_INNER * 2);        // tanh(C)
    __bf16* yh    = (__bf16*)carve((size_t)MTOT * D_INNER * 2);        // gated y bf16
    float*  Pbuf  = (float*)carve((size_t)BATCH * NC * D_INNER * 4);
    float*  Qbuf  = (float*)carve((size_t)BATCH * NC * D_INNER * 4);
    float*  S0buf = (float*)carve((size_t)BATCH * NC * D_INNER * 4);

    const int TB = 256;
    auto cvt = [&](const float* s, __bf16* d, int n) {
        cvt_bf16_kernel<<<(n + TB - 1) / TB, TB, 0, stream>>>(s, d, n);
    };
    cvt(x,     xh,  MTOT * D_MODEL);
    cvt(w_in,  w1h, 2 * D_INNER * D_MODEL);
    cvt(dt_w,  wdh,                       D_INNER * D_INNER);
    cvt(b_w,   wdh + (size_t)1 * D_INNER * D_INNER, D_INNER * D_INNER);
    cvt(c_w,   wdh + (size_t)2 * D_INNER * D_INNER, D_INNER * D_INNER);
    cvt(out_w, woh, D_MODEL * D_INNER);

    // GEMM 1: xz = x @ in_proj_w^T + b   [16384 x 4096]
    gemm_wmma<0><<<dim3((2 * D_INNER) / 128, MTOT / 128), TB, 0, stream>>>(
        xh, w1h, D_MODEL, 2 * D_INNER, b_in,
        nullptr, nullptr, xzh, nullptr, nullptr, nullptr, nullptr);

    // depthwise conv + SiLU -> u
    conv_silu_kernel<<<dim3(D_INNER / TB, MTOT), TB, 0, stream>>>(xzh, conv_w, conv_b, uh);

    // GEMM 2 (fused dt|B|C): [16384 x 6144], epilogue -> decay, tanh(B), tanh(C)
    gemm_wmma<1><<<dim3((3 * D_INNER) / 128, MTOT / 128), TB, 0, stream>>>(
        uh, wdh, D_INNER, 3 * D_INNER, nullptr,
        dt_b, A_log, nullptr, nullptr, decayB, Bh, Ch);

    // chunked scan + fused gate epilogue
    scan_pass1<<<(BATCH * NC * D_INNER) / TB, TB, 0, stream>>>(decayB, Bh, uh, Pbuf, Qbuf);
    scan_pass2<<<(BATCH * D_INNER) / TB, TB, 0, stream>>>(Pbuf, Qbuf, S0buf);
    scan_pass3<<<(BATCH * NC * D_INNER) / TB, TB, 0, stream>>>(
        decayB, Bh, uh, Ch, xzh, D_par, S0buf, yh);

    // GEMM 3: out = y @ out_w^T + out_b  [16384 x 1024] f32
    gemm_wmma<2><<<dim3(D_MODEL / 128, MTOT / 128), TB, 0, stream>>>(
        yh, woh, D_INNER, D_MODEL, out_b,
        nullptr, nullptr, nullptr, out, nullptr, nullptr, nullptr);
}
